// AttentionBlock_22273700397498
// MI455X (gfx1250) — compile-verified, matrix operand order checked by reading, not run on hardware
//
#include <hip/hip_runtime.h>
#include <math.h>
#include <stdint.h>

typedef _Float16 h16;
typedef __attribute__((ext_vector_type(16))) _Float16 v16h;
typedef __attribute__((ext_vector_type(8)))  _Float16 v8h;
typedef __attribute__((ext_vector_type(8)))  float    v8f;

#define B_  16
#define C_  512
#define N_  1024
#define HN_ 8
#define D_  64
#define O3_ 1536

// Gather a 16x32 fragment given the lane's base pointer (row already applied).
// Elements 0..7 at p[0..7], elements 8..15 at p[16..23] (K +16).
__device__ __forceinline__ v16h frag_at(const h16* p) {
    v8h a = *(const v8h*)p;
    v8h b = *(const v8h*)(p + 16);
    v16h f;
#pragma unroll
    for (int i = 0; i < 8; ++i) { f[i] = a[i]; f[i+8] = b[i]; }
    return f;
}

// Row-major global fragment: lane L: row = row0 + (L&15), K base = k0 + 8*(L>>4)
__device__ __forceinline__ v16h load_frag(const h16* __restrict__ base, int row0, int k0, int ldm) {
    int lane = threadIdx.x & 31;
    return frag_at(base + (size_t)(row0 + (lane & 15)) * ldm + k0 + 8 * (lane >> 4));
}

__device__ __forceinline__ v8f wmma32(v16h a, v16h b, v8f c) {
    return __builtin_amdgcn_wmma_f32_16x16x32_f16(false, a, false, b, (short)0, c, false, false);
}

// ---------------- Kernel 1: forced weight normalization -> f16 ----------------
__global__ void k_wnorm(const float* __restrict__ wp, const float* __restrict__ wo,
                        h16* __restrict__ wp_h, h16* __restrict__ wo_h) {
    int wave = (blockIdx.x * blockDim.x + threadIdx.x) >> 5;
    int lane = threadIdx.x & 31;
    const float* src; h16* dst;
    if (wave < O3_)            { src = wp + (size_t)wave * C_; dst = wp_h + (size_t)wave * C_; }
    else if (wave < O3_ + C_)  { int r2 = wave - O3_;
                                 src = wo + (size_t)r2 * C_;   dst = wo_h + (size_t)r2 * C_; }
    else return;
    float vals[16]; float ssq = 0.f;
#pragma unroll
    for (int i = 0; i < 16; ++i) { float v = src[lane + 32*i]; vals[i] = v; ssq += v*v; }
#pragma unroll
    for (int off = 16; off > 0; off >>= 1) ssq += __shfl_xor(ssq, off, 32);
    float n = sqrtf(ssq);
    const float alpha   = 0.04419417382f;   // sqrt(1/512)
    const float inv_fan = 0.04419417382f;   // 1/sqrt(512)
    float s = inv_fan / (1e-4f + alpha * n);
#pragma unroll
    for (int i = 0; i < 16; ++i) dst[lane + 32*i] = (h16)(vals[i] * s);
}

// ---------------- Kernel 2: LDS-tiled transpose x [B,C,N] -> xt [B,N,C] f16 ----------------
__global__ void k_transpose(const float* __restrict__ x, h16* __restrict__ xt) {
    __shared__ float tile[32][33];
    int b  = blockIdx.z;
    int n0 = blockIdx.x * 32, c0 = blockIdx.y * 32;
    int tx = threadIdx.x & 31, ty8 = threadIdx.x >> 5;
#pragma unroll
    for (int i = 0; i < 4; ++i) {
        int c = ty8 + 8*i;
        tile[c][tx] = x[((size_t)b*C_ + c0 + c)*N_ + n0 + tx];
    }
    __syncthreads();
#pragma unroll
    for (int i = 0; i < 4; ++i) {
        int n = ty8 + 8*i;
        xt[((size_t)b*N_ + n0 + n)*C_ + c0 + tx] = (h16)tile[tx][n];
    }
}

// ---------------- Kernel 3: QKV projection GEMM (WMMA, 16x64 per wave) ----------------
// h[b][n][o] = sum_c xt[b][n][c] * wp[o][c]
__global__ void k_qkv_gemm(const h16* __restrict__ xt, const h16* __restrict__ wp_h,
                           h16* __restrict__ qkv) {
    __shared__ alignas(16) h16 stage[8][16 * 72];   // wave-local 16x64 tile, row stride 72
    int w = threadIdx.x >> 5, lane = threadIdx.x & 31;
    int otile0 = (blockIdx.x * 8 + w) * 64;
    int mtile  = blockIdx.y * 16;
    int b = blockIdx.z;
    const h16* A = xt + (size_t)b * N_ * C_;
    v8f a0 = {}, a1 = {}, a2 = {}, a3 = {};
    for (int c0 = 0; c0 < C_; c0 += 32) {
        v16h af = load_frag(A, mtile, c0, C_);
        a0 = wmma32(af, load_frag(wp_h, otile0 + 0,  c0, C_), a0);
        a1 = wmma32(af, load_frag(wp_h, otile0 + 16, c0, C_), a1);
        a2 = wmma32(af, load_frag(wp_h, otile0 + 32, c0, C_), a2);
        a3 = wmma32(af, load_frag(wp_h, otile0 + 48, c0, C_), a3);
    }
    int lr = lane & 15, hi = lane >> 4;
    h16* sp = &stage[w][0];
#pragma unroll
    for (int r = 0; r < 8; ++r) {
        int m = r + 8*hi;
        sp[m*72 + 0  + lr] = (h16)a0[r];
        sp[m*72 + 16 + lr] = (h16)a1[r];
        sp[m*72 + 32 + lr] = (h16)a2[r];
        sp[m*72 + 48 + lr] = (h16)a3[r];
    }
    // wave-local staging: no barrier needed
    h16* outp = qkv + ((size_t)b*N_ + mtile) * O3_ + otile0;
#pragma unroll
    for (int i = 0; i < 4; ++i) {
        int chunk = lane + 32*i;               // 0..127
        int row = chunk >> 3, c8 = chunk & 7;
        v8h v = *(const v8h*)&sp[row*72 + c8*8];
        *(v8h*)(outp + (size_t)row*O3_ + c8*8) = v;
    }
}

// ---------------- Kernel 4: pixnorm + scale + head scatter ----------------
__global__ void k_pixnorm(const h16* __restrict__ qkv, h16* __restrict__ q,
                          h16* __restrict__ k, h16* __restrict__ vT) {
    int tid = blockIdx.x * blockDim.x + threadIdx.x;  // ((b*HN + h)*N + n): n fastest
    int n = tid & (N_ - 1);
    int h = (tid >> 10) & (HN_ - 1);
    int b = tid >> 13;
    if (b >= B_) return;
    const h16* src = qkv + ((size_t)b * N_ + n) * O3_ + h * (3 * D_);
    v8h bq[8], bk[8], bv[8];
    float sq = 0.f, sk = 0.f, sv = 0.f;
#pragma unroll
    for (int i = 0; i < 8; ++i) {
        bq[i] = *(const v8h*)(src + i*8);
        bk[i] = *(const v8h*)(src + 64 + i*8);
        bv[i] = *(const v8h*)(src + 128 + i*8);
#pragma unroll
        for (int j = 0; j < 8; ++j) {
            float a = (float)bq[i][j]; sq += a*a;
            float c = (float)bk[i][j]; sk += c*c;
            float d = (float)bv[i][j]; sv += d*d;
        }
    }
    const float scale = 0.35355339059f;                   // d_k^{-1/4}
    float rq = scale * rsqrtf(sq * (1.f/64.f) + 1e-8f);
    float rk = scale * rsqrtf(sk * (1.f/64.f) + 1e-8f);
    float rv =         rsqrtf(sv * (1.f/64.f) + 1e-8f);
    size_t head = (size_t)b * HN_ + h;
    h16* qd = q  + (head * N_ + n) * (size_t)D_;
    h16* kd = k  + (head * N_ + n) * (size_t)D_;
    h16* vd = vT + head * (size_t)D_ * N_ + n;            // d-major; n coalesced across threads
#pragma unroll
    for (int i = 0; i < 8; ++i) {
        v8h oq, ok;
#pragma unroll
        for (int j = 0; j < 8; ++j) { oq[j] = (h16)((float)bq[i][j]*rq); ok[j] = (h16)((float)bk[i][j]*rk); }
        *(v8h*)(qd + i*8) = oq;
        *(v8h*)(kd + i*8) = ok;
#pragma unroll
        for (int j = 0; j < 8; ++j) vd[(size_t)(i*8 + j) * N_] = (h16)((float)bv[i][j]*rv);
    }
}

// ---------------- Kernel 5: transposed flash attention (WMMA + async LDS pipeline) ----
// Block = 8 waves sharing one (b,head); K/V tiles double-buffered in LDS via
// global_load_async_to_lds_b128 (ASYNCcnt). S^T = K*Q^T so softmax axis is the
// register axis; P fragments feed the PV WMMA directly (no transpose).
__global__ void __launch_bounds__(256) k_attn(const h16* __restrict__ q, const h16* __restrict__ k,
                                              const h16* __restrict__ vT, h16* __restrict__ att) {
    __shared__ alignas(16) h16 sK[2][32 * 64];   // 32 keys x 64 d
    __shared__ alignas(16) h16 sV[2][64 * 32];   // 64 d x 32 keys (d-major)
    int w = threadIdx.x >> 5, lane = threadIdx.x & 31;
    int bh = blockIdx.x >> 3;                    // b*HN + h  (uniform per block)
    int qt = (blockIdx.x & 7) * 8 + w;
    int b  = bh >> 3, h = bh & (HN_ - 1);
    const h16* qb = q  + (size_t)bh * N_ * D_;
    const h16* kb = k  + (size_t)bh * N_ * D_;
    const h16* vb = vT + (size_t)bh * D_ * N_;

    uint32_t t   = threadIdx.x;
    uint32_t kgo = ((t >> 3) * 64u + (t & 7u) * 8u) * 2u;            // K: 32 rows x 128B
    uint32_t vgo = ((t >> 2) * (uint32_t)N_ + (t & 3u) * 8u) * 2u;   // V: 64 rows x stride 2KB
    uint32_t vlo = ((t >> 2) * 32u + (t & 3u) * 8u) * 2u;
    uint32_t sKb = (uint32_t)(uintptr_t)&sK[0][0];
    uint32_t sVb = (uint32_t)(uintptr_t)&sV[0][0];
    unsigned long long kaddr = (unsigned long long)(uintptr_t)kb;
    unsigned long long vaddr = (unsigned long long)(uintptr_t)vb;

    auto issue = [&](uint32_t jtn, uint32_t p) {
        uint32_t ldsK = sKb + p * (32u*64u*2u) + kgo;
        uint32_t gK   = kgo + jtn * 128u;
        asm volatile("global_load_async_to_lds_b128 %0, %1, %2"
                     :: "v"(ldsK), "v"(gK), "s"(kaddr) : "memory");
        uint32_t ldsV = sVb + p * (64u*32u*2u) + vlo;
        uint32_t gV   = vgo + jtn * 2u;
        asm volatile("global_load_async_to_lds_b128 %0, %1, %2"
                     :: "v"(ldsV), "v"(gV), "s"(vaddr) : "memory");
    };

    v16h qf0 = load_frag(qb, qt * 16, 0,  D_);
    v16h qf1 = load_frag(qb, qt * 16, 32, D_);
    v8f o0 = {}, o1 = {}, o2 = {}, o3 = {};
    float m_run = -1e30f, l_run = 0.f;
    int lr = lane & 15, hi = lane >> 4;

    issue(0u, 0u);
    for (int jt = 0; jt < N_; jt += 32) {
        uint32_t p = (uint32_t)(jt >> 5) & 1u;
        issue((uint32_t)((jt + 32) & (N_ - 1)), p ^ 1u);   // prefetch next tile
        asm volatile("s_wait_asynccnt 0x2" ::: "memory");  // current tile landed
        __syncthreads();
        const h16* kp = &sK[p][0];
        const h16* vp = &sV[p][0];

        v8f s0 = {}, s1 = {};
        s0 = wmma32(frag_at(kp + lr*64 +      8*hi), qf0, s0);
        s0 = wmma32(frag_at(kp + lr*64 + 32 + 8*hi), qf1, s0);
        s1 = wmma32(frag_at(kp + (16+lr)*64 +      8*hi), qf0, s1);
        s1 = wmma32(frag_at(kp + (16+lr)*64 + 32 + 8*hi), qf1, s1);

        float mloc = s0[0];
#pragma unroll
        for (int r = 1; r < 8; ++r) mloc = fmaxf(mloc, s0[r]);
#pragma unroll
        for (int r = 0; r < 8; ++r) mloc = fmaxf(mloc, s1[r]);
        mloc = fmaxf(mloc, __shfl_xor(mloc, 16, 32));      // share max with query partner lane
        float mnew = fmaxf(m_run, mloc);
        float corr = __expf(m_run - mnew);
        m_run = mnew;
        l_run *= corr;
#pragma unroll
        for (int r = 0; r < 8; ++r) { o0[r]*=corr; o1[r]*=corr; o2[r]*=corr; o3[r]*=corr; }
        v16h pf;
        float lsum = 0.f;
#pragma unroll
        for (int r = 0; r < 8; ++r) { float pv = __expf(s0[r]-mnew); lsum += pv; pf[r]     = (h16)pv; }
#pragma unroll
        for (int r = 0; r < 8; ++r) { float pv = __expf(s1[r]-mnew); lsum += pv; pf[8 + r] = (h16)pv; }
        l_run += lsum;

        o0 = wmma32(frag_at(vp + (0*16 + lr)*32 + 8*hi), pf, o0);
        o1 = wmma32(frag_at(vp + (1*16 + lr)*32 + 8*hi), pf, o1);
        o2 = wmma32(frag_at(vp + (2*16 + lr)*32 + 8*hi), pf, o2);
        o3 = wmma32(frag_at(vp + (3*16 + lr)*32 + 8*hi), pf, o3);
        __syncthreads();    // all reads of buffer p done before next iter overwrites it
    }

    float l_tot = l_run + __shfl_xor(l_run, 16, 32);
    float inv = 1.f / l_tot;
    int n = qt * 16 + lr;
    h16* dst = att + ((size_t)b * N_ + n) * C_ + h * D_ + 8 * hi;
    v8h o;
#pragma unroll
    for (int r = 0; r < 8; ++r) o[r] = (h16)(o0[r] * inv);
    *(v8h*)(dst + 0)  = o;
#pragma unroll
    for (int r = 0; r < 8; ++r) o[r] = (h16)(o1[r] * inv);
    *(v8h*)(dst + 16) = o;
#pragma unroll
    for (int r = 0; r < 8; ++r) o[r] = (h16)(o2[r] * inv);
    *(v8h*)(dst + 32) = o;
#pragma unroll
    for (int r = 0; r < 8; ++r) o[r] = (h16)(o3[r] * inv);
    *(v8h*)(dst + 48) = o;
}

// ---------------- Kernel 6: output GEMM (16x64/wave) + magnitude-preserving sum ----------
__global__ void k_out_gemm(const h16* __restrict__ att, const h16* __restrict__ wo_h,
                           const float* __restrict__ x, float* __restrict__ out) {
    int w = threadIdx.x >> 5, lane = threadIdx.x & 31;
    int otile = (blockIdx.y * 8 + w) * 16;
    int nt0   = blockIdx.x * 64;
    int b = blockIdx.z;
    const h16* Bm = att + (size_t)b * N_ * C_;
    v8f a0 = {}, a1 = {}, a2 = {}, a3 = {};
    for (int c0 = 0; c0 < C_; c0 += 32) {
        v16h af = load_frag(wo_h, otile, c0, C_);
        a0 = wmma32(af, load_frag(Bm, nt0 + 0,  c0, C_), a0);
        a1 = wmma32(af, load_frag(Bm, nt0 + 16, c0, C_), a1);
        a2 = wmma32(af, load_frag(Bm, nt0 + 32, c0, C_), a2);
        a3 = wmma32(af, load_frag(Bm, nt0 + 48, c0, C_), a3);
    }
    int lr = lane & 15, hi = lane >> 4;
    const float ax = 0.91914503f;   // 0.7 / sqrt(0.58)
    const float bx = 0.39391930f;   // 0.3 / sqrt(0.58)
#pragma unroll
    for (int r = 0; r < 8; ++r) {
        size_t rowb = ((size_t)b * C_ + otile + r + 8*hi) * N_;
        out[rowb + nt0 + 0  + lr] = ax * x[rowb + nt0 + 0  + lr] + bx * a0[r];
        out[rowb + nt0 + 16 + lr] = ax * x[rowb + nt0 + 16 + lr] + bx * a1[r];
        out[rowb + nt0 + 32 + lr] = ax * x[rowb + nt0 + 32 + lr] + bx * a2[r];
        out[rowb + nt0 + 48 + lr] = ax * x[rowb + nt0 + 48 + lr] + bx * a3[r];
    }
}

extern "C" void kernel_launch(void* const* d_in, const int* in_sizes, int n_in,
                              void* d_out, int out_size, void* d_ws, size_t ws_size,
                              hipStream_t stream) {
    const float* x  = (const float*)d_in[0];
    const float* wp = (const float*)d_in[1];
    const float* wo = (const float*)d_in[2];
    float* out = (float*)d_out;
    char* ws = (char*)d_ws;

    h16* wp_h = (h16*)(ws + 0);                  //  1.5 MB
    h16* wo_h = (h16*)(ws + 1572864);            //  0.5 MB
    h16* xt   = (h16*)(ws + 2097152);            // 16 MB (reused as att)
    h16* qkv  = (h16*)(ws + 18874368);           // 48 MB
    h16* qh   = (h16*)(ws + 69206016);           // 16 MB
    h16* kh   = (h16*)(ws + 85983232);           // 16 MB
    h16* vT   = (h16*)(ws + 102760448);          // 16 MB (total ~114 MB)
    h16* att  = xt;                              // xt dead after kernel 3

    k_wnorm    <<<256, 256, 0, stream>>>(wp, wo, wp_h, wo_h);
    k_transpose<<<dim3(N_/32, C_/32, B_), 256, 0, stream>>>(x, xt);
    k_qkv_gemm <<<dim3(O3_/512, N_/16, B_), 256, 0, stream>>>(xt, wp_h, qkv);
    k_pixnorm  <<<(B_*HN_*N_)/256, 256, 0, stream>>>(qkv, qh, kh, vT);
    k_attn     <<<B_*HN_*8, 256, 0, stream>>>(qh, kh, vT, att);
    k_out_gemm <<<dim3(N_/64, C_/128, B_), 256, 0, stream>>>(att, wo_h, x, out);
}
